// TopKSwiGLUFF_28656021799426
// MI455X (gfx1250) — matrix operand order repaired and checked by reading, not compiled, not run on hardware
//
#include <hip/hip_runtime.h>
#include <hip/hip_bf16.h>

#define D_MODEL 2048
#define D_FF    8192
#define N_TOK   8192
#define TOP_K   2048

typedef __bf16 v16bf __attribute__((ext_vector_type(16)));
typedef __bf16 v8bf  __attribute__((ext_vector_type(8)));
typedef float  v8f   __attribute__((ext_vector_type(8)));
typedef unsigned short v8us __attribute__((ext_vector_type(8)));

// 128-bit payload type for the async-to-LDS builtin (gcc-style vector, per
// the builtin's prototype: int __attribute__((vector_size(16))))
typedef int v4i __attribute__((vector_size(16)));
typedef __attribute__((address_space(1))) v4i* as1_v4i_p;   // global
typedef __attribute__((address_space(3))) v4i* as3_v4i_p;   // LDS

union FragBF { v16bf v; v8bf h[2]; };

#if __has_builtin(__builtin_amdgcn_global_load_async_to_lds_b128)
#define HAVE_ASYNC_LDS 1
#endif

static __device__ __forceinline__ void wait_async0() {
#if __has_builtin(__builtin_amdgcn_s_wait_asynccnt)
  __builtin_amdgcn_s_wait_asynccnt(0);
#else
  asm volatile("s_wait_asynccnt 0" ::: "memory");
#endif
}

static __device__ __forceinline__ unsigned short f2bf(float f) {
  unsigned u = __float_as_uint(f);
  u += 0x7FFFu + ((u >> 16) & 1u);   // round-to-nearest-even
  return (unsigned short)(u >> 16);
}

// ---------------------------------------------------------------------------
// fp32 -> bf16 bulk convert (vectorized, grid-stride)
// ---------------------------------------------------------------------------
__global__ __launch_bounds__(256) void cvt_f32_bf16(const float* __restrict__ s,
                                                    unsigned short* __restrict__ d,
                                                    int n4) {
  int i = blockIdx.x * blockDim.x + threadIdx.x;
  int stride = gridDim.x * blockDim.x;
  for (; i < n4; i += stride) {
    float4 v = ((const float4*)s)[i];
    ushort4 o;
    o.x = f2bf(v.x); o.y = f2bf(v.y); o.z = f2bf(v.z); o.w = f2bf(v.w);
    ((ushort4*)d)[i] = o;
  }
}

// ---------------------------------------------------------------------------
// GEMM1 fused: gate[t,f] = silu(x@w1^T) * (x@w3^T), bf16 out.
// Block tile 128(M) x 64(N); 8 waves -> 32x32 each via 2x2 WMMA 16x16x32 bf16.
// A (x) double-buffered in LDS via async-to-LDS (ASYNCcnt) when available;
// B (w1/w3) streamed from global as 32B contiguous per-lane fragments.
// One barrier per K-step (copy of next tile overlaps current WMMA work).
// ---------------------------------------------------------------------------
__global__ __launch_bounds__(256) void gemm1_swiglu(
    const unsigned short* __restrict__ xb,
    const unsigned short* __restrict__ w1b,
    const unsigned short* __restrict__ w3b,
    unsigned short* __restrict__ gate) {
  __shared__ unsigned char ldsA[2][128 * 80];

  const int tid  = threadIdx.x;
  const int lane = tid & 31;
  const int wave = tid >> 5;
  const int wm   = wave >> 1;      // 0..3  -> M sub-tile
  const int wn   = wave & 1;       // 0..1  -> N sub-tile
  const int mblock = blockIdx.y * 128;
  const int fblock = blockIdx.x * 64;

  const v8f zero = {0.f,0.f,0.f,0.f,0.f,0.f,0.f,0.f};
  v8f acc1[2][2], acc3[2][2];
#pragma unroll
  for (int mi = 0; mi < 2; ++mi)
#pragma unroll
    for (int ni = 0; ni < 2; ++ni) { acc1[mi][ni] = zero; acc3[mi][ni] = zero; }

  // B fragment base pointers: lane -> (n = lane&15, k-group = (lane>>4)*16)
  const int nIdx  = lane & 15;
  const int kpart = (lane >> 4) * 16;
  const unsigned short* b1p[2];
  const unsigned short* b3p[2];
#pragma unroll
  for (int ni = 0; ni < 2; ++ni) {
    int f = fblock + wn * 32 + ni * 16 + nIdx;
    b1p[ni] = w1b + (size_t)f * D_MODEL + kpart;
    b3p[ni] = w3b + (size_t)f * D_MODEL + kpart;
  }

  // A staging: thread -> (row = tid>>1, 16-elt half = tid&1)
  const int ldr = tid >> 1, ldh = tid & 1;
  const unsigned short* gsrc = xb + (size_t)(mblock + ldr) * D_MODEL + ldh * 16;
  const int stoff = ldr * 80 + ldh * 32;

  int rowoff[2];
#pragma unroll
  for (int mi = 0; mi < 2; ++mi)
    rowoff[mi] = (wm * 32 + mi * 16 + (lane & 15)) * 80 + (lane >> 4) * 32;

  auto issue = [&](int k0, int buf) {
    unsigned char* lp = &ldsA[buf][stoff];
    const unsigned short* gp = gsrc + k0;
#ifdef HAVE_ASYNC_LDS
    __builtin_amdgcn_global_load_async_to_lds_b128(
        (as1_v4i_p)(const void*)gp, (as3_v4i_p)(void*)lp, 0, 0);
    __builtin_amdgcn_global_load_async_to_lds_b128(
        (as1_v4i_p)(const void*)gp, (as3_v4i_p)(void*)lp, 16, 0);
#else
    ((v8us*)lp)[0] = ((const v8us*)gp)[0];
    ((v8us*)lp)[1] = ((const v8us*)gp)[1];
#endif
  };

  issue(0, 0);
  int buf = 0;
  for (int k0 = 0; k0 < D_MODEL; k0 += 32, buf ^= 1) {
#ifdef HAVE_ASYNC_LDS
    wait_async0();
#endif
    __syncthreads();
    if (k0 + 32 < D_MODEL) issue(k0 + 32, buf ^ 1);

    __builtin_prefetch((const void*)(b1p[0] + k0 + 64), 0, 3);
    __builtin_prefetch((const void*)(b3p[0] + k0 + 64), 0, 3);

    FragBF a[2];
#pragma unroll
    for (int mi = 0; mi < 2; ++mi) {
      const unsigned char* ap = &ldsA[buf][rowoff[mi]];
      a[mi].h[0] = *(const v8bf*)(ap);
      a[mi].h[1] = *(const v8bf*)(ap + 16);
    }
#pragma unroll
    for (int ni = 0; ni < 2; ++ni) {
      FragBF b1, b3;
      b1.v = *(const v16bf*)(b1p[ni] + k0);
      b3.v = *(const v16bf*)(b3p[ni] + k0);
#pragma unroll
      for (int mi = 0; mi < 2; ++mi) {
        acc1[mi][ni] = __builtin_amdgcn_wmma_f32_16x16x32_bf16(
            false, a[mi].v, false, b1.v, (short)0, acc1[mi][ni], false, false);
        acc3[mi][ni] = __builtin_amdgcn_wmma_f32_16x16x32_bf16(
            false, a[mi].v, false, b3.v, (short)0, acc3[mi][ni], false, false);
      }
    }
  }

  // Epilogue: C/D layout -> elem i: M = i + 8*(lane>>4), N = lane&15
  const int mrow0 = mblock + wm * 32 + 8 * (lane >> 4);
  const int ncol0 = fblock + wn * 32 + (lane & 15);
#pragma unroll
  for (int mi = 0; mi < 2; ++mi)
#pragma unroll
    for (int ni = 0; ni < 2; ++ni)
#pragma unroll
      for (int i = 0; i < 8; ++i) {
        float h1 = acc1[mi][ni][i];
        float h3 = acc3[mi][ni][i];
        float g  = (h1 / (1.0f + __expf(-h1))) * h3;
        gate[(size_t)(mrow0 + mi * 16 + i) * D_FF + (ncol0 + ni * 16)] = f2bf(g);
      }
}

// ---------------------------------------------------------------------------
// Exact per-token k-th largest |gate| via 2-pass radix select on bf16 |bits|.
// Writes only the 16-bit threshold key per token (no gate rewrite; GEMM2
// applies the mask during staging). Same tie semantics as reference.
// ---------------------------------------------------------------------------
__global__ __launch_bounds__(256) void topk_thresh(const unsigned short* __restrict__ gate,
                                                   unsigned short* __restrict__ thr) {
  __shared__ unsigned int hist[256];
  __shared__ unsigned int s_hi, s_rank;

  const int tid = threadIdx.x;
  const unsigned short* rp = gate + (size_t)blockIdx.x * D_FF + tid * 32;

  v8us raw[4];
#pragma unroll
  for (int j = 0; j < 4; ++j) raw[j] = ((const v8us*)rp)[j];

  unsigned short keys[32];
#pragma unroll
  for (int j = 0; j < 4; ++j)
#pragma unroll
    for (int e = 0; e < 8; ++e)
      keys[j * 8 + e] = (unsigned short)(raw[j][e] & 0x7FFFu);

  // pass 1: histogram of high 7 bits
  hist[tid] = 0;
  __syncthreads();
#pragma unroll
  for (int j = 0; j < 32; ++j) atomicAdd(&hist[keys[j] >> 8], 1u);
  __syncthreads();
  if (tid == 0) {
    unsigned cum = 0, hi = 0, rank = TOP_K;
    for (int b = 127; b >= 0; --b) {
      if (cum + hist[b] >= (unsigned)TOP_K) { hi = (unsigned)b; rank = TOP_K - cum; break; }
      cum += hist[b];
    }
    s_hi = hi; s_rank = rank;
  }
  __syncthreads();
  const unsigned hi = s_hi, rank = s_rank;

  // pass 2: histogram of low 8 bits within the selected high bucket
  hist[tid] = 0;
  __syncthreads();
#pragma unroll
  for (int j = 0; j < 32; ++j)
    if ((unsigned)(keys[j] >> 8) == hi) atomicAdd(&hist[keys[j] & 0xFFu], 1u);
  __syncthreads();
  if (tid == 0) {
    unsigned cum = 0, th = hi << 8;
    for (int b = 255; b >= 0; --b) {
      cum += hist[b];
      if (cum >= rank) { th = (hi << 8) | (unsigned)b; break; }
    }
    thr[blockIdx.x] = (unsigned short)th;
  }
}

// ---------------------------------------------------------------------------
// GEMM2: out[t,d] = sum_f mask(gate[t,f]) * w2[d,f], f32 out.
// Top-k mask applied in registers while staging gate tiles into LDS
// (threshold per row preloaded to LDS). Double-buffered, 1 barrier/K-step.
// ---------------------------------------------------------------------------
__global__ __launch_bounds__(256) void gemm2(
    const unsigned short* __restrict__ gb,
    const unsigned short* __restrict__ w2b,
    const unsigned short* __restrict__ thr,
    float* __restrict__ out) {
  __shared__ unsigned char ldsA[2][128 * 80];
  __shared__ unsigned short thbuf[128];

  const int tid  = threadIdx.x;
  const int lane = tid & 31;
  const int wave = tid >> 5;
  const int wm   = wave >> 1;
  const int wn   = wave & 1;
  const int mblock = blockIdx.y * 128;
  const int dblock = blockIdx.x * 64;

  if (tid < 128) thbuf[tid] = thr[mblock + tid];
  __syncthreads();

  const v8f zero = {0.f,0.f,0.f,0.f,0.f,0.f,0.f,0.f};
  v8f acc[2][2];
#pragma unroll
  for (int mi = 0; mi < 2; ++mi)
#pragma unroll
    for (int ni = 0; ni < 2; ++ni) acc[mi][ni] = zero;

  const int nIdx  = lane & 15;
  const int kpart = (lane >> 4) * 16;
  const unsigned short* bp[2];
#pragma unroll
  for (int ni = 0; ni < 2; ++ni) {
    int d = dblock + wn * 32 + ni * 16 + nIdx;
    bp[ni] = w2b + (size_t)d * D_FF + kpart;
  }

  const int ldr = tid >> 1, ldh = tid & 1;
  const unsigned short* gsrc = gb + (size_t)(mblock + ldr) * D_FF + ldh * 16;
  const int stoff = ldr * 80 + ldh * 32;
  const unsigned short trow = thbuf[ldr];

  int rowoff[2];
#pragma unroll
  for (int mi = 0; mi < 2; ++mi)
    rowoff[mi] = (wm * 32 + mi * 16 + (lane & 15)) * 80 + (lane >> 4) * 32;

  // masked staging: zero entries whose |bits| key is below the row threshold
  auto issue = [&](int k0, int bufi) {
    const unsigned short* gp = gsrc + k0;
    v8us r0 = ((const v8us*)gp)[0];
    v8us r1 = ((const v8us*)gp)[1];
#pragma unroll
    for (int e = 0; e < 8; ++e) {
      if ((unsigned short)(r0[e] & 0x7FFFu) < trow) r0[e] = (unsigned short)0;
      if ((unsigned short)(r1[e] & 0x7FFFu) < trow) r1[e] = (unsigned short)0;
    }
    unsigned char* lp = &ldsA[bufi][stoff];
    ((v8us*)lp)[0] = r0;
    ((v8us*)lp)[1] = r1;
  };

  issue(0, 0);
  int buf = 0;
  for (int k0 = 0; k0 < D_FF; k0 += 32, buf ^= 1) {
    __syncthreads();
    if (k0 + 32 < D_FF) issue(k0 + 32, buf ^ 1);

    __builtin_prefetch((const void*)(bp[0] + k0 + 64), 0, 3);

    FragBF a[2];
#pragma unroll
    for (int mi = 0; mi < 2; ++mi) {
      const unsigned char* ap = &ldsA[buf][rowoff[mi]];
      a[mi].h[0] = *(const v8bf*)(ap);
      a[mi].h[1] = *(const v8bf*)(ap + 16);
    }
#pragma unroll
    for (int ni = 0; ni < 2; ++ni) {
      FragBF b;
      b.v = *(const v16bf*)(bp[ni] + k0);
#pragma unroll
      for (int mi = 0; mi < 2; ++mi) {
        acc[mi][ni] = __builtin_amdgcn_wmma_f32_16x16x32_bf16(
            false, a[mi].v, false, b.v, (short)0, acc[mi][ni], false, false);
      }
    }
  }

  const int mrow0 = mblock + wm * 32 + 8 * (lane >> 4);
  const int ncol0 = dblock + wn * 32 + (lane & 15);
#pragma unroll
  for (int mi = 0; mi < 2; ++mi)
#pragma unroll
    for (int ni = 0; ni < 2; ++ni)
#pragma unroll
      for (int i = 0; i < 8; ++i)
        out[(size_t)(mrow0 + mi * 16 + i) * D_MODEL + (ncol0 + ni * 16)] = acc[mi][ni][i];
}

// ---------------------------------------------------------------------------
extern "C" void kernel_launch(void* const* d_in, const int* in_sizes, int n_in,
                              void* d_out, int out_size, void* d_ws, size_t ws_size,
                              hipStream_t stream) {
  const float* x  = (const float*)d_in[0];
  const float* w1 = (const float*)d_in[1];
  const float* w2 = (const float*)d_in[2];
  const float* w3 = (const float*)d_in[3];

  unsigned short* ws  = (unsigned short*)d_ws;
  const size_t nX = (size_t)N_TOK * D_MODEL;   // 16,777,216 elems
  const size_t nW = (size_t)D_FF * D_MODEL;    // 16,777,216 elems

  unsigned short* xb   = ws;
  unsigned short* w1b  = xb  + nX;
  unsigned short* w3b  = w1b + nW;
  unsigned short* w2b  = w3b + nW;
  unsigned short* gate = w2b + nW;             // N_TOK * D_FF bf16 (128 MB)
  unsigned short* thr  = gate + (size_t)N_TOK * D_FF;  // per-token threshold keys

  cvt_f32_bf16<<<2048, 256, 0, stream>>>(x,  xb,  (int)(nX / 4));
  cvt_f32_bf16<<<2048, 256, 0, stream>>>(w1, w1b, (int)(nW / 4));
  cvt_f32_bf16<<<2048, 256, 0, stream>>>(w3, w3b, (int)(nW / 4));
  cvt_f32_bf16<<<2048, 256, 0, stream>>>(w2, w2b, (int)(nW / 4));

  gemm1_swiglu<<<dim3(D_FF / 64, N_TOK / 128), 256, 0, stream>>>(xb, w1b, w3b, gate);
  topk_thresh<<<N_TOK, 256, 0, stream>>>(gate, thr);
  gemm2<<<dim3(D_MODEL / 64, N_TOK / 128), 256, 0, stream>>>(gate, w2b, thr, (float*)d_out);
}
